// Attention_59983513256145
// MI455X (gfx1250) — 8-bit matrix operand order repaired and checked by reading, compiled, not run on hardware
//
#include <hip/hip_runtime.h>
#include <hip/hip_bf16.h>
#include <stdint.h>


typedef __attribute__((ext_vector_type(8))) int v8i;

#define B_  8
#define N_  1024
#define C_  768
#define H_  12
#define HD_ 64
#define C3_ 2304
#define M_  (B_ * N_)
#define QMAX 127.0f

__device__ __forceinline__ v8i frag_a(const signed char* base, int lda, int k0, int lane) {
    const int m = lane & 15, h = lane >> 4;
    const signed char* rp = base + (size_t)m * lda + k0 + (h << 3);
    v8i f;
#pragma unroll
    for (int v = 0; v < 8; ++v) {
        const int kb = ((v >> 1) << 4) + ((v & 1) << 2);
        f[v] = *(const int*)(rp + kb);
    }
    return f;
}

__device__ __forceinline__ v8i frag_b(const signed char* baseT, int ldb, int k0, int lane) {
    const int n = lane & 15, h = lane >> 4;
    const signed char* cp = baseT + (size_t)n * ldb + k0 + (h << 3);
    v8i f;
#pragma unroll
    for (int v = 0; v < 8; ++v) {
        const int kb = ((v >> 1) << 4) + ((v & 1) << 2);
        f[v] = *(const int*)(cp + kb);
    }
    return f;
}

__device__ __forceinline__ float q_scale(const unsigned* amax) {
    return fmaxf(__uint_as_float(*amax) * (1.0f / QMAX), 1e-8f);
}
__device__ __forceinline__ int q8(float x, float inv_s) {
    float r = rintf(x * inv_s);
    r = fminf(fmaxf(r, -128.0f), 127.0f);
    return (int)r;
}
__device__ __forceinline__ int pack4(int a, int b, int c, int d) {
    return (a & 255) | ((b & 255) << 8) | ((c & 255) << 16) | ((d & 255) << 24);
}

__global__ void init_amax_kernel(unsigned* amax) {
    if (threadIdx.x < 8) amax[threadIdx.x] = 0u;
}

__global__ void amax_kernel(const float* __restrict__ x, size_t n, unsigned* out) {
    __shared__ float red[256];
    float m = 0.0f;
    for (size_t i = (size_t)blockIdx.x * blockDim.x + threadIdx.x; i < n;
         i += (size_t)gridDim.x * blockDim.x)
        m = fmaxf(m, fabsf(x[i]));
    red[threadIdx.x] = m;
    __syncthreads();
    for (int s = 128; s > 0; s >>= 1) {
        if (threadIdx.x < s) red[threadIdx.x] = fmaxf(red[threadIdx.x], red[threadIdx.x + s]);
        __syncthreads();
    }
    if (threadIdx.x == 0) atomicMax(out, __float_as_uint(red[0]));
}

__global__ void amax_cols_kernel(const float* __restrict__ qkv, int col0, unsigned* out) {
    __shared__ float red[256];
    const size_t n = (size_t)M_ * C_;
    float m = 0.0f;
    for (size_t i = (size_t)blockIdx.x * blockDim.x + threadIdx.x; i < n;
         i += (size_t)gridDim.x * blockDim.x) {
        size_t row = i / C_;
        int c = (int)(i % C_);
        m = fmaxf(m, fabsf(qkv[row * C3_ + col0 + c]));
    }
    red[threadIdx.x] = m;
    __syncthreads();
    for (int s = 128; s > 0; s >>= 1) {
        if (threadIdx.x < s) red[threadIdx.x] = fmaxf(red[threadIdx.x], red[threadIdx.x + s]);
        __syncthreads();
    }
    if (threadIdx.x == 0) atomicMax(out, __float_as_uint(red[0]));
}

__global__ void quant_i8_kernel(const float* __restrict__ x, const unsigned* amax,
                                signed char* __restrict__ y, size_t n4) {
    size_t i = (size_t)blockIdx.x * blockDim.x + threadIdx.x;
    if (i >= n4) return;
    const float inv_s = 1.0f / q_scale(amax);
    const float4 v = ((const float4*)x)[i];
    ((int*)y)[i] = pack4(q8(v.x, inv_s), q8(v.y, inv_s), q8(v.z, inv_s), q8(v.w, inv_s));
}

__global__ void quant_tr_kernel(const float* __restrict__ w, const unsigned* amax,
                                signed char* __restrict__ wt, int K, int Ncol) {
    size_t i = (size_t)blockIdx.x * blockDim.x + threadIdx.x;
    const size_t total = (size_t)K * Ncol;
    if (i >= total) return;
    const float inv_s = 1.0f / q_scale(amax);
    int k = (int)(i / Ncol), j = (int)(i % Ncol);
    wt[(size_t)j * K + k] = (signed char)q8(w[i], inv_s);
}

__global__ void quant_qk_kernel(const float* __restrict__ qkv, const unsigned* amax,
                                signed char* __restrict__ dst, int col0) {
    size_t t = (size_t)blockIdx.x * blockDim.x + threadIdx.x;
    if (t >= (size_t)M_ * C_) return;
    const float inv_s = 1.0f / q_scale(amax);
    size_t row = t / C_;
    int c = (int)(t % C_);
    int hh = c >> 6, d = c & 63;
    int b = (int)(row >> 10), n = (int)(row & 1023);
    float v = qkv[row * C3_ + col0 + c];
    dst[((((size_t)b * H_ + hh) * N_ + n) << 6) + d] = (signed char)q8(v, inv_s);
}

__global__ void quant_v_t_kernel(const float* __restrict__ qkv, const unsigned* amax,
                                 signed char* __restrict__ dst) {
    size_t t = (size_t)blockIdx.x * blockDim.x + threadIdx.x;
    if (t >= (size_t)M_ * C_) return;
    const float inv_s = 1.0f / q_scale(amax);
    size_t row = t / C_;
    int c = (int)(t % C_);
    int hh = c >> 6, d = c & 63;
    int b = (int)(row >> 10), n = (int)(row & 1023);
    float v = qkv[row * C3_ + 2 * C_ + c];
    dst[((((size_t)b * H_ + hh) * HD_ + d) << 10) + n] = (signed char)q8(v, inv_s);
}

__global__ __launch_bounds__(256) void gemm_i8_kernel(
    const signed char* __restrict__ A, const signed char* __restrict__ Bt,
    const float* __restrict__ bias, const unsigned* amaxA, const unsigned* amaxB,
    float* __restrict__ Cout, int M, int N, int K) {
    const int lane = threadIdx.x & 31, wave = threadIdx.x >> 5;
    const int row0 = blockIdx.y * 128 + (wave >> 1) * 32;
    const int col0 = blockIdx.x * 128 + (wave & 1) * 64;
    v8i acc[2][4];
#pragma unroll
    for (int i = 0; i < 2; ++i)
#pragma unroll
        for (int j = 0; j < 4; ++j) acc[i][j] = (v8i){};
    for (int k0 = 0; k0 < K; k0 += 64) {
        v8i a0 = frag_a(A + (size_t)row0 * K, K, k0, lane);
        v8i a1 = frag_a(A + (size_t)(row0 + 16) * K, K, k0, lane);
        v8i b[4];
#pragma unroll
        for (int j = 0; j < 4; ++j)
            b[j] = frag_b(Bt + (size_t)(col0 + 16 * j) * K, K, k0, lane);
#pragma unroll
        for (int j = 0; j < 4; ++j) {
            acc[0][j] = __builtin_amdgcn_wmma_i32_16x16x64_iu8(true, a0, true, b[j], acc[0][j], false, false);
            acc[1][j] = __builtin_amdgcn_wmma_i32_16x16x64_iu8(true, a1, true, b[j], acc[1][j], false, false);
        }
    }
    const float ds = q_scale(amaxA) * q_scale(amaxB);
    const int h = lane >> 4, n = lane & 15;
#pragma unroll
    for (int j = 0; j < 4; ++j) {
        const float bia = bias[col0 + 16 * j + n];
#pragma unroll
        for (int r = 0; r < 8; ++r) {
            const int m = h * 8 + r;
            Cout[(size_t)(row0 + m) * N + col0 + 16 * j + n]      = (float)acc[0][j][r] * ds + bia;
            Cout[(size_t)(row0 + 16 + m) * N + col0 + 16 * j + n] = (float)acc[1][j][r] * ds + bia;
        }
    }
}

__global__ __launch_bounds__(256) void attn_stats_kernel(
    const signed char* __restrict__ qq, const signed char* __restrict__ kq,
    const unsigned* amax_q, const unsigned* amax_k,
    float* __restrict__ m_arr, float* __restrict__ l_arr, unsigned* amax_p) {
    const int lane = threadIdx.x & 31, wave = threadIdx.x >> 5;
    const int wb = blockIdx.x * 8 + wave;
    const int bh = wb >> 6, rblk = wb & 63;
    const signed char* qbase = qq + ((size_t)bh * N_ + rblk * 16) * HD_;
    const signed char* kbase = kq + (size_t)bh * N_ * HD_;
    const float sqk = q_scale(amax_q) * q_scale(amax_k) * 0.125f;
    const v8i a = frag_a(qbase, HD_, 0, lane);
    float mloc[8], lloc[8];
#pragma unroll
    for (int r = 0; r < 8; ++r) { mloc[r] = -3.4e38f; lloc[r] = 0.0f; }
    for (int t = 0; t < N_ / 16; ++t) {
        v8i bq = frag_b(kbase + (size_t)t * 16 * HD_, HD_, 0, lane);
        v8i c = {};
        c = __builtin_amdgcn_wmma_i32_16x16x64_iu8(true, a, true, bq, c, false, false);
#pragma unroll
        for (int r = 0; r < 8; ++r) {
            const float s = (float)c[r] * sqk;
            const float nm = fmaxf(mloc[r], s);
            lloc[r] = lloc[r] * __expf(mloc[r] - nm) + __expf(s - nm);
            mloc[r] = nm;
        }
    }
    float M[8], L[8];
#pragma unroll
    for (int r = 0; r < 8; ++r) {
        float mr = mloc[r];
#pragma unroll
        for (int msk = 1; msk < 16; msk <<= 1) mr = fmaxf(mr, __shfl_xor(mr, msk, 32));
        float lr = lloc[r] * __expf(mloc[r] - mr);
#pragma unroll
        for (int msk = 1; msk < 16; msk <<= 1) lr += __shfl_xor(lr, msk, 32);
        M[r] = mr;
        L[r] = lr;
    }
    const int h = lane >> 4, n = lane & 15;
    if (n == 0) {
        const int rowbase = bh * N_ + rblk * 16 + h * 8;
        float pm = 0.0f;
#pragma unroll
        for (int r = 0; r < 8; ++r) {
            m_arr[rowbase + r] = M[r];
            l_arr[rowbase + r] = L[r];
            pm = fmaxf(pm, 1.0f / L[r]);
        }
        atomicMax(amax_p, __float_as_uint(pm));
    }
}

__global__ __launch_bounds__(256) void attn_out_kernel(
    const signed char* __restrict__ qq, const signed char* __restrict__ kq,
    const signed char* __restrict__ vqt, const float* __restrict__ m_arr,
    const float* __restrict__ l_arr, const unsigned* amax_q, const unsigned* amax_k,
    const unsigned* amax_p, const unsigned* amax_v, float* __restrict__ out) {
    __shared__ uint8_t pbuf[8 * 16 * 64];
    const int lane = threadIdx.x & 31, wave = threadIdx.x >> 5;
    const int wb = blockIdx.x * 8 + wave;
    const int bh = wb >> 6, rblk = wb & 63;
    const int b = bh / H_, hh = bh % H_;
    const signed char* qbase = qq + ((size_t)bh * N_ + rblk * 16) * HD_;
    const signed char* kbase = kq + (size_t)bh * N_ * HD_;
    const signed char* vbase = vqt + (size_t)bh * HD_ * N_;
    const float sqk = q_scale(amax_q) * q_scale(amax_k) * 0.125f;
    const float sp = q_scale(amax_p);
    const float dso = sp * q_scale(amax_v);
    const v8i aq = frag_a(qbase, HD_, 0, lane);
    const int h = lane >> 4, n = lane & 15;
    const int rowbase = bh * N_ + rblk * 16 + h * 8;
    float Mr[8], Ci[8];
#pragma unroll
    for (int r = 0; r < 8; ++r) {
        Mr[r] = m_arr[rowbase + r];
        Ci[r] = 1.0f / (l_arr[rowbase + r] * sp);
    }
    v8i o0 = {}, o1 = {}, o2 = {}, o3 = {};
    uint8_t* myp = pbuf + wave * (16 * 64);
    for (int kt = 0; kt < N_ / 64; ++kt) {
        const int j0 = kt * 64;
#pragma unroll
        for (int i = 0; i < 4; ++i) {
            v8i bq = frag_b(kbase + (size_t)(j0 + i * 16) * HD_, HD_, 0, lane);
            v8i c = {};
            c = __builtin_amdgcn_wmma_i32_16x16x64_iu8(true, aq, true, bq, c, false, false);
#pragma unroll
            for (int r = 0; r < 8; ++r) {
                const float s = (float)c[r] * sqk;
                int pq = (int)rintf(__expf(s - Mr[r]) * Ci[r]);
                pq = pq < 0 ? 0 : (pq > 127 ? 127 : pq);
                myp[(h * 8 + r) * 64 + i * 16 + n] = (uint8_t)pq;
            }
        }
        asm volatile("s_wait_dscnt 0" ::: "memory");
        const v8i ap = frag_a((const signed char*)myp, 64, 0, lane);
        v8i b0 = frag_b(vbase + (size_t)0 * 16 * N_, N_, j0, lane);
        v8i b1 = frag_b(vbase + (size_t)1 * 16 * N_, N_, j0, lane);
        v8i b2 = frag_b(vbase + (size_t)2 * 16 * N_, N_, j0, lane);
        v8i b3 = frag_b(vbase + (size_t)3 * 16 * N_, N_, j0, lane);
        o0 = __builtin_amdgcn_wmma_i32_16x16x64_iu8(false, ap, true, b0, o0, false, false);
        o1 = __builtin_amdgcn_wmma_i32_16x16x64_iu8(false, ap, true, b1, o1, false, false);
        o2 = __builtin_amdgcn_wmma_i32_16x16x64_iu8(false, ap, true, b2, o2, false, false);
        o3 = __builtin_amdgcn_wmma_i32_16x16x64_iu8(false, ap, true, b3, o3, false, false);
        asm volatile("s_wait_dscnt 0" ::: "memory");
    }
#pragma unroll
    for (int r = 0; r < 8; ++r) {
        const int row = rblk * 16 + h * 8 + r;
        float* op = out + ((size_t)b * N_ + row) * C_ + hh * HD_ + n;
        op[0]  = (float)o0[r] * dso;
        op[16] = (float)o1[r] * dso;
        op[32] = (float)o2[r] * dso;
        op[48] = (float)o3[r] * dso;
    }
}

extern "C" void kernel_launch(void* const* d_in, const int* in_sizes, int n_in,
                              void* d_out, int out_size, void* d_ws, size_t ws_size,
                              hipStream_t stream) {
    const float* x      = (const float*)d_in[0];
    const float* w_qkv  = (const float*)d_in[1];
    const float* b_qkv  = (const float*)d_in[2];
    const float* w_proj = (const float*)d_in[3];
    const float* b_proj = (const float*)d_in[4];
    float* out = (float*)d_out;

    char* ws = (char*)d_ws;
    size_t off = 0;
    auto take = [&](size_t bytes) { size_t o = off; off = (off + bytes + 255) & ~(size_t)255; return o; };
    unsigned*    amax   = (unsigned*)(ws + take(8 * sizeof(unsigned)));
    signed char* xq     = (signed char*)(ws + take((size_t)M_ * C_));
    signed char* wqkvT  = (signed char*)(ws + take((size_t)C3_ * C_));
    signed char* wprojT = (signed char*)(ws + take((size_t)C_ * C_));
    float*       qkv    = (float*)(ws + take((size_t)M_ * C3_ * 4));
    signed char* qq     = (signed char*)(ws + take((size_t)M_ * C_));
    signed char* kq     = (signed char*)(ws + take((size_t)M_ * C_));
    signed char* vqt    = (signed char*)(ws + take((size_t)M_ * C_));
    float*       m_arr  = (float*)(ws + take((size_t)B_ * H_ * N_ * 4));
    float*       l_arr  = (float*)(ws + take((size_t)B_ * H_ * N_ * 4));
    float*       ao     = (float*)(ws + take((size_t)M_ * C_ * 4));
    signed char* aoq    = (signed char*)(ws + take((size_t)M_ * C_));

    const size_t nx = (size_t)M_ * C_;
    const size_t nwqkv = (size_t)C_ * C3_;
    const size_t nwproj = (size_t)C_ * C_;

    hipLaunchKernelGGL(init_amax_kernel, dim3(1), dim3(64), 0, stream, amax);
    hipLaunchKernelGGL(amax_kernel, dim3(2048), dim3(256), 0, stream, x, nx, amax + 0);
    hipLaunchKernelGGL(amax_kernel, dim3(1024), dim3(256), 0, stream, w_qkv, nwqkv, amax + 1);
    hipLaunchKernelGGL(amax_kernel, dim3(512), dim3(256), 0, stream, w_proj, nwproj, amax + 2);
    hipLaunchKernelGGL(quant_i8_kernel, dim3((unsigned)((nx / 4 + 255) / 256)), dim3(256), 0,
                       stream, x, amax + 0, xq, nx / 4);
    hipLaunchKernelGGL(quant_tr_kernel, dim3((unsigned)((nwqkv + 255) / 256)), dim3(256), 0,
                       stream, w_qkv, amax + 1, wqkvT, C_, C3_);
    hipLaunchKernelGGL(quant_tr_kernel, dim3((unsigned)((nwproj + 255) / 256)), dim3(256), 0,
                       stream, w_proj, amax + 2, wprojT, C_, C_);
    hipLaunchKernelGGL(gemm_i8_kernel, dim3(C3_ / 128, M_ / 128), dim3(256), 0, stream,
                       xq, wqkvT, b_qkv, amax + 0, amax + 1, qkv, M_, C3_, C_);
    hipLaunchKernelGGL(amax_cols_kernel, dim3(2048), dim3(256), 0, stream, qkv, 0, amax + 3);
    hipLaunchKernelGGL(amax_cols_kernel, dim3(2048), dim3(256), 0, stream, qkv, C_, amax + 4);
    hipLaunchKernelGGL(amax_cols_kernel, dim3(2048), dim3(256), 0, stream, qkv, 2 * C_, amax + 5);
    hipLaunchKernelGGL(quant_qk_kernel, dim3((unsigned)((nx + 255) / 256)), dim3(256), 0, stream,
                       qkv, amax + 3, qq, 0);
    hipLaunchKernelGGL(quant_qk_kernel, dim3((unsigned)((nx + 255) / 256)), dim3(256), 0, stream,
                       qkv, amax + 4, kq, C_);
    hipLaunchKernelGGL(quant_v_t_kernel, dim3((unsigned)((nx + 255) / 256)), dim3(256), 0, stream,
                       qkv, amax + 5, vqt);
    const unsigned ablocks = (unsigned)(B_ * H_ * (N_ / 16) / 8);
    hipLaunchKernelGGL(attn_stats_kernel, dim3(ablocks), dim3(256), 0, stream,
                       qq, kq, amax + 3, amax + 4, m_arr, l_arr, amax + 6);
    hipLaunchKernelGGL(attn_out_kernel, dim3(ablocks), dim3(256), 0, stream,
                       qq, kq, vqt, m_arr, l_arr, amax + 3, amax + 4, amax + 6, amax + 5, ao);
    hipLaunchKernelGGL(amax_kernel, dim3(2048), dim3(256), 0, stream, ao, nx, amax + 7);
    hipLaunchKernelGGL(quant_i8_kernel, dim3((unsigned)((nx / 4 + 255) / 256)), dim3(256), 0,
                       stream, ao, amax + 7, aoq, nx / 4);
    hipLaunchKernelGGL(gemm_i8_kernel, dim3(C_ / 128, M_ / 128), dim3(256), 0, stream,
                       aoq, wprojT, b_proj, amax + 7, amax + 2, out, M_, C_, C_);
    (void)in_sizes; (void)n_in; (void)out_size; (void)ws_size;
}
